// BodyOrderModel_13503377179220
// MI455X (gfx1250) — matrix operand order repaired and checked by reading, not compiled, not run on hardware
//
#include <hip/hip_runtime.h>
#include <hip/hip_bf16.h>

typedef __attribute__((ext_vector_type(16))) __bf16       v16bf;
typedef __attribute__((ext_vector_type(8)))  float        v8f;
typedef __attribute__((ext_vector_type(4)))  unsigned int v4u;

#define CDIM       128
#define A_IN_DIM   10
#define NGRAPH     512
#define PI_F       3.14159265358979f
// sqrt(3), sqrt(15), sqrt(5/8), sqrt(3/8)
#define S3_F   1.7320508075688772f
#define S15_F  3.872983346207417f
#define CA_F   0.7905694150420949f
#define CB_F   0.6123724356957945f

// ---------------------------------------------------------------------------
// Async memory->LDS copy (gfx1250 GLOBAL_LOAD_ASYNC_TO_LDS_B128, ASYNCcnt).
// LDS byte offset = low 32 bits of the flat shared-memory address (ISA 10.2).
// ---------------------------------------------------------------------------
__device__ __forceinline__ void async_copy_b128(unsigned lds_off,
                                                const void* gaddr) {
  asm volatile("global_load_async_to_lds_b128 %0, %1, off"
               :
               : "v"(lds_off), "v"(gaddr)
               : "memory");
}
__device__ __forceinline__ void wait_async_zero() {
  asm volatile("s_wait_asynccnt 0x0" ::: "memory");
}

// ---------------------------------------------------------------------------
// Convert + transpose W_edge / W_msg (3 x 128 x 128, [l][k][n]) to bf16 [l][n][k]
// so WMMA B-fragments become k-contiguous 16-byte LDS reads.
// ---------------------------------------------------------------------------
__global__ void prep_weights_kernel(const float* __restrict__ W_edge,
                                    const float* __restrict__ W_msg,
                                    __bf16* __restrict__ wedgeT,
                                    __bf16* __restrict__ wmsgT) {
  int idx = blockIdx.x * blockDim.x + threadIdx.x;
  if (idx >= 3 * CDIM * CDIM) return;
  int l   = idx >> 14;             // / 16384
  int rem = idx & (CDIM * CDIM - 1);
  int k   = rem >> 7;
  int n   = rem & (CDIM - 1);
  int dst = (l * CDIM + n) * CDIM + k;
  wedgeT[dst] = (__bf16)W_edge[idx];
  wmsgT[dst]  = (__bf16)W_msg[idx];
}

// ---------------------------------------------------------------------------
// node_feats = one_hot @ W_embed ; energies[0][batch[n]] += attrs . atomic_E
// ---------------------------------------------------------------------------
__global__ void embed_kernel(const float* __restrict__ node_attrs,
                             const float* __restrict__ W_embed,
                             const float* __restrict__ atomic_energies,
                             const int*   __restrict__ batch,
                             float* __restrict__ nf_out,
                             float* __restrict__ energies,
                             int N) {
  int n = blockIdx.x;
  int c = threadIdx.x;
  if (n >= N) return;
  float attrs[A_IN_DIM];
#pragma unroll
  for (int a = 0; a < A_IN_DIM; ++a) attrs[a] = node_attrs[n * A_IN_DIM + a];
  float acc = 0.f;
#pragma unroll
  for (int a = 0; a < A_IN_DIM; ++a) acc += attrs[a] * W_embed[a * CDIM + c];
  nf_out[n * CDIM + c] = acc;
  if (c == 0) {
    float e0 = 0.f;
#pragma unroll
    for (int a = 0; a < A_IN_DIM; ++a) e0 += attrs[a] * atomic_energies[a];
    atomicAdd(&energies[batch[n]], e0);
  }
}

// ---------------------------------------------------------------------------
// Fused edge kernel: async-stage W_edge^T into LDS while recomputing edge
// features (bessel x sph harmonics) into an LDS bf16 tile, WMMA bf16 GEMM,
// multiply by gathered node_feats[sender], scatter-add into agg[receiver].
// Block = 256 threads (8 wave32) over 128 edges.
// ---------------------------------------------------------------------------
__global__ __launch_bounds__(256) void edge_kernel(
    const float* __restrict__ edge_vectors,
    const float* __restrict__ edge_lengths,
    const __bf16* __restrict__ wT,       // [128][128] (n-major) for this layer
    const int*   __restrict__ sender,
    const int*   __restrict__ receiver,
    const float* __restrict__ nf_in,
    float* __restrict__ agg,
    int E) {
  __shared__ __align__(16) __bf16 wlds[CDIM * CDIM];   // 32 KB
  __shared__ __align__(16) __bf16 ef[128 * CDIM];      // 32 KB edge-feature tile
  __shared__ int s_send[128];
  __shared__ int s_recv[128];

  int t = threadIdx.x;

  // Kick off async staging of W^T (32 KB = 2048 x 16B); overlaps with the
  // transcendental-heavy edge-feature computation below.
  {
    const char* g    = (const char*)wT;
    unsigned    ldsW = (unsigned)(size_t)(const void*)wlds;
#pragma unroll
    for (int i = 0; i < 8; ++i) {
      unsigned off = (unsigned)(t + i * 256) * 16u;
      async_copy_b128(ldsW + off, g + off);
    }
  }

  // Compute the 128x128 bf16 edge-feature tile (threads 0..127, one edge each)
  int eBase = blockIdx.x * 128;
  if (t < 128) {
    int e = eBase + t;
    unsigned int* efu = (unsigned int*)&ef[t * CDIM];
    if (e < E) {
      float r   = edge_lengths[e];
      float inv = 1.f / r;
      float x = edge_vectors[e * 3 + 0] * inv;
      float y = edge_vectors[e * 3 + 1] * inv;
      float z = edge_vectors[e * 3 + 2] * inv;
      float xx = x * x, yy = y * y, zz = z * z;
      float sph[16];
      sph[0]  = 1.f;
      sph[1]  = x;
      sph[2]  = y;
      sph[3]  = z;
      sph[4]  = S3_F * x * y;
      sph[5]  = S3_F * y * z;
      sph[6]  = 0.5f * (3.f * zz - 1.f);
      sph[7]  = S3_F * x * z;
      sph[8]  = 0.5f * S3_F * (xx - yy);
      sph[9]  = CA_F * y * (3.f * xx - yy);
      sph[10] = S15_F * x * y * z;
      sph[11] = CB_F * y * (5.f * zz - 1.f);
      sph[12] = 0.5f * z * (5.f * zz - 3.f);
      sph[13] = CB_F * x * (5.f * zz - 1.f);
      sph[14] = 0.5f * S15_F * z * (xx - yy);
      sph[15] = CA_F * x * (xx - 3.f * yy);

      float u  = r * 0.2f;                 // r / R_MAX
      float u2 = u * u, u3 = u2 * u, u6 = u3 * u3, u7 = u6 * u, u8 = u7 * u;
      float cut = 1.f - 28.f * u6 + 48.f * u7 - 21.f * u8;
      cut = (u < 1.f) ? cut : 0.f;
      float pref = 0.6324555320336759f * inv * cut;   // sqrt(2/R_MAX)/r * cut
      float bes[8];
#pragma unroll
      for (int b = 0; b < 8; ++b) bes[b] = pref * __sinf((float)(b + 1) * PI_F * u);

#pragma unroll
      for (int b = 0; b < 8; ++b) {
#pragma unroll
        for (int s = 0; s < 16; s += 2) {
          __bf16 v0 = (__bf16)(bes[b] * sph[s]);
          __bf16 v1 = (__bf16)(bes[b] * sph[s + 1]);
          efu[(b * 16 + s) >> 1] =
              (unsigned int)__builtin_bit_cast(unsigned short, v0) |
              ((unsigned int)__builtin_bit_cast(unsigned short, v1) << 16);
        }
      }
      s_send[t] = sender[e] * CDIM;
      s_recv[t] = receiver[e] * CDIM;
    } else {
#pragma unroll
      for (int i = 0; i < 64; ++i) efu[i] = 0u;
      s_send[t] = 0;
      s_recv[t] = 0;
    }
  }
  wait_async_zero();
  __syncthreads();

  int lane = t & 31;
  int w    = t >> 5;
  int lo   = lane & 15;
  int hi   = lane >> 4;
  int mB   = w * 16;                       // 16-edge tile per wave

  // Hoist the wave's sender/receiver offsets (tile-invariant) into registers.
  int soff[8], roff[8];
#pragma unroll
  for (int v = 0; v < 8; ++v) {
    soff[v] = s_send[mB + hi * 8 + v];
    roff[v] = s_recv[mB + hi * 8 + v];
  }

  // A fragments for K = 0..127 (4 chunks), cached across all 8 N-tiles.
  // ISA layout 16x32 bf16: lane=row; VGPR0-3 = K k0+hi*8..+7, VGPR4-7 = +16.
  v16bf afrag[4];
#pragma unroll
  for (int kc = 0; kc < 4; ++kc) {
    const v4u* pa = (const v4u*)&ef[(mB + lo) * CDIM + kc * 32 + hi * 8];
    ((v4u*)&afrag[kc])[0] = pa[0];
    ((v4u*)&afrag[kc])[1] = pa[2];         // +32 bytes = +16 bf16
  }

#pragma unroll
  for (int nt = 0; nt < 8; ++nt) {
    v8f acc = {};
#pragma unroll
    for (int kc = 0; kc < 4; ++kc) {
      // B layout 32x16 bf16: lane=col; K = kc*32+hi*8..+7 and +16..+23
      const v4u* pb = (const v4u*)&wlds[(nt * 16 + lo) * CDIM + kc * 32 + hi * 8];
      v16bf bfrag;
      ((v4u*)&bfrag)[0] = pb[0];
      ((v4u*)&bfrag)[1] = pb[2];
      acc = __builtin_amdgcn_wmma_f32_16x16x32_bf16(
          false, afrag[kc], false, bfrag, (short)0, acc, false, false);
    }
    int col = nt * 16 + lo;
    // Batch the 8 gathers so their latency overlaps, then scatter.
    float nfv[8];
#pragma unroll
    for (int v = 0; v < 8; ++v) nfv[v] = nf_in[soff[v] + col];
#pragma unroll
    for (int v = 0; v < 8; ++v)
      atomicAdd(&agg[roff[v] + col], acc[v] * nfv[v]);
  }
}

// ---------------------------------------------------------------------------
// Node kernel: nf_out = agg @ W_msg + nf_in * (attrs @ W_skip) ; per-graph
// readout energies accumulated via LDS then one global atomic per node.
// ---------------------------------------------------------------------------
__global__ __launch_bounds__(256) void node_kernel(
    const float* __restrict__ agg,
    const __bf16* __restrict__ wmsgT,     // [128][128] (n-major) layer slice
    const float* __restrict__ node_attrs,
    const float* __restrict__ W_skip,     // [A_IN][128] layer slice
    const float* __restrict__ W_readout,  // [128] layer slice
    const int*   __restrict__ batch,
    const float* __restrict__ nf_in,
    float* __restrict__ nf_out,
    float* __restrict__ energies,         // slice for layer l+1
    int N) {
  __shared__ __align__(16) __bf16 wlds[CDIM * CDIM];   // 32 KB
  __shared__ __align__(16) __bf16 al[128 * CDIM];      // 32 KB agg tile (bf16)
  __shared__ float s_attr[128 * A_IN_DIM];
  __shared__ float s_wskip[A_IN_DIM * CDIM];
  __shared__ float s_wread[CDIM];
  __shared__ float s_ne[128];

  int t     = threadIdx.x;
  int nBase = blockIdx.x * 128;

  // Async-stage W_msg^T; overlaps with the fp32->bf16 agg conversion below.
  {
    const char* g    = (const char*)wmsgT;
    unsigned    ldsW = (unsigned)(size_t)(const void*)wlds;
#pragma unroll
    for (int i = 0; i < 8; ++i) {
      unsigned off = (unsigned)(t + i * 256) * 16u;
      async_copy_b128(ldsW + off, g + off);
    }
  }
  {
    unsigned int* au = (unsigned int*)al;
#pragma unroll
    for (int i = 0; i < 32; ++i) {
      int ui  = t + i * 256;               // 0..8191 dword pairs
      int row = ui >> 6;
      int cp  = (ui & 63) * 2;
      int gn  = nBase + row;
      float v0 = 0.f, v1 = 0.f;
      if (gn < N) {
        v0 = agg[gn * CDIM + cp];
        v1 = agg[gn * CDIM + cp + 1];
      }
      au[ui] = (unsigned int)__builtin_bit_cast(unsigned short, (__bf16)v0) |
               ((unsigned int)__builtin_bit_cast(unsigned short, (__bf16)v1) << 16);
    }
  }
  for (int i = t; i < A_IN_DIM * CDIM; i += 256) s_wskip[i] = W_skip[i];
  if (t < CDIM) s_wread[t] = W_readout[t];
  if (t < 128) {
    s_ne[t] = 0.f;
    int gn = nBase + t;
#pragma unroll
    for (int a = 0; a < A_IN_DIM; ++a)
      s_attr[t * A_IN_DIM + a] = (gn < N) ? node_attrs[gn * A_IN_DIM + a] : 0.f;
  }
  wait_async_zero();
  __syncthreads();

  int lane = t & 31, w = t >> 5, lo = lane & 15, hi = lane >> 4;
  int mB = w * 16;

  v16bf afrag[4];
#pragma unroll
  for (int kc = 0; kc < 4; ++kc) {
    const v4u* pa = (const v4u*)&al[(mB + lo) * CDIM + kc * 32 + hi * 8];
    ((v4u*)&afrag[kc])[0] = pa[0];
    ((v4u*)&afrag[kc])[1] = pa[2];
  }

#pragma unroll
  for (int nt = 0; nt < 8; ++nt) {
    v8f acc = {};
#pragma unroll
    for (int kc = 0; kc < 4; ++kc) {
      const v4u* pb = (const v4u*)&wlds[(nt * 16 + lo) * CDIM + kc * 32 + hi * 8];
      v16bf bfrag;
      ((v4u*)&bfrag)[0] = pb[0];
      ((v4u*)&bfrag)[1] = pb[2];
      acc = __builtin_amdgcn_wmma_f32_16x16x32_bf16(
          false, afrag[kc], false, bfrag, (short)0, acc, false, false);
    }
    int col = nt * 16 + lo;
#pragma unroll
    for (int v = 0; v < 8; ++v) {
      int nl = mB + hi * 8 + v;
      int gn = nBase + nl;
      if (gn < N) {
        float skipw = 0.f;
#pragma unroll
        for (int a = 0; a < A_IN_DIM; ++a)
          skipw += s_attr[nl * A_IN_DIM + a] * s_wskip[a * CDIM + col];
        float nv = acc[v] + nf_in[gn * CDIM + col] * skipw;
        nf_out[gn * CDIM + col] = nv;
        atomicAdd(&s_ne[nl], nv * s_wread[col]);   // ds_add_f32
      }
    }
  }
  __syncthreads();
  if (t < 128) {
    int gn = nBase + t;
    if (gn < N) atomicAdd(&energies[batch[gn]], s_ne[t]);
  }
}

// ---------------------------------------------------------------------------
// d_out = [ total_energy(512) | energies(4x512) ]
// ---------------------------------------------------------------------------
__global__ void finalize_kernel(const float* __restrict__ energies,
                                float* __restrict__ out) {
  int g = blockIdx.x * blockDim.x + threadIdx.x;
  if (g >= NGRAPH) return;
  float tot = 0.f;
#pragma unroll
  for (int l = 0; l < 4; ++l) {
    float v = energies[l * NGRAPH + g];
    out[NGRAPH + l * NGRAPH + g] = v;
    tot += v;
  }
  out[g] = tot;
}

extern "C" void kernel_launch(void* const* d_in, const int* in_sizes, int n_in,
                              void* d_out, int out_size, void* d_ws, size_t ws_size,
                              hipStream_t stream) {
  const float* node_attrs      = (const float*)d_in[0];
  const float* edge_vectors    = (const float*)d_in[1];
  const float* edge_lengths    = (const float*)d_in[2];
  const float* W_embed         = (const float*)d_in[3];
  const float* atomic_energies = (const float*)d_in[4];
  const float* W_edge          = (const float*)d_in[5];
  const float* W_msg           = (const float*)d_in[6];
  const float* W_skip          = (const float*)d_in[7];
  const float* W_readout       = (const float*)d_in[8];
  const int*   edge_index      = (const int*)d_in[9];
  const int*   batch           = (const int*)d_in[10];

  const int N = in_sizes[0] / A_IN_DIM;
  const int E = in_sizes[2];
  const int* sender   = edge_index;
  const int* receiver = edge_index + E;

  char*  ws  = (char*)d_ws;
  size_t off = 0;
  auto alloc = [&](size_t bytes) -> void* {
    void* p = ws + off;
    off += (bytes + 255) & ~(size_t)255;
    return p;
  };
  size_t  nfBytes  = (size_t)N * CDIM * sizeof(float);
  float*  nfA      = (float*)alloc(nfBytes);
  float*  nfB      = (float*)alloc(nfBytes);
  float*  agg      = (float*)alloc(nfBytes);
  __bf16* wedgeT   = (__bf16*)alloc(3 * CDIM * CDIM * sizeof(__bf16));
  __bf16* wmsgT    = (__bf16*)alloc(3 * CDIM * CDIM * sizeof(__bf16));
  float*  energies = (float*)alloc(4 * NGRAPH * sizeof(float));
  (void)n_in; (void)out_size; (void)ws_size;

  hipMemsetAsync(energies, 0, 4 * NGRAPH * sizeof(float), stream);
  prep_weights_kernel<<<(3 * CDIM * CDIM + 255) / 256, 256, 0, stream>>>(
      W_edge, W_msg, wedgeT, wmsgT);
  embed_kernel<<<N, CDIM, 0, stream>>>(node_attrs, W_embed, atomic_energies,
                                       batch, nfA, energies, N);

  float* nin  = nfA;
  float* nout = nfB;
  int eBlocks = (E + 127) / 128;
  int nBlocks = (N + 127) / 128;
  for (int l = 0; l < 3; ++l) {
    hipMemsetAsync(agg, 0, nfBytes, stream);
    edge_kernel<<<eBlocks, 256, 0, stream>>>(
        edge_vectors, edge_lengths, wedgeT + (size_t)l * CDIM * CDIM,
        sender, receiver, nin, agg, E);
    node_kernel<<<nBlocks, 256, 0, stream>>>(
        agg, wmsgT + (size_t)l * CDIM * CDIM, node_attrs,
        W_skip + (size_t)l * A_IN_DIM * CDIM, W_readout + (size_t)l * CDIM,
        batch, nin, nout, energies + (size_t)(l + 1) * NGRAPH, N);
    float* tmp = nin; nin = nout; nout = tmp;
  }
  finalize_kernel<<<(NGRAPH + 255) / 256, 256, 0, stream>>>(energies,
                                                            (float*)d_out);
}
